// MultiHeadAttention_12060268167522
// MI455X (gfx1250) — matrix operand order repaired and checked by reading, not compiled, not run on hardware
//
#include <hip/hip_runtime.h>
#include <hip/hip_bf16.h>

// Problem constants (reference: B=4, T=2048, C=1024, H=16, DK=64)
#define BATCH 4
#define TSEQ  2048
#define CDIM  1024
#define NHEAD 16
#define DK    64
#define MROWS (BATCH * TSEQ)       // 8192
#define F3    (3 * CDIM)           // 3072

typedef __attribute__((ext_vector_type(16))) __bf16 v16bf;
typedef __attribute__((ext_vector_type(8)))  float  v8f;

union Frag {
  v16bf v;
  uint4 q[2];
};

__device__ __forceinline__ v8f wmma_bf16(const Frag& a, const Frag& b, v8f c) {
  return __builtin_amdgcn_wmma_f32_16x16x32_bf16(false, a.v, false, b.v,
                                                 (short)0, c, false, false);
}

// ---------------------------------------------------------------------------
// f32 -> bf16 conversion
// ---------------------------------------------------------------------------
__global__ void f32_to_bf16_kernel(const float* __restrict__ in,
                                   __bf16* __restrict__ out, int n) {
  int i = blockIdx.x * blockDim.x + threadIdx.x;
  if (i < n) out[i] = (__bf16)in[i];
}

// ---------------------------------------------------------------------------
// Gather V into [bh][d][t] (transposed) so PV B-fragments are contiguous.
// ---------------------------------------------------------------------------
__global__ void vt_gather_kernel(const __bf16* __restrict__ qkv,
                                 __bf16* __restrict__ vt) {
  int idx = blockIdx.x * blockDim.x + threadIdx.x;
  const int total = BATCH * NHEAD * DK * TSEQ;
  if (idx >= total) return;
  int t  = idx & (TSEQ - 1);
  int d  = (idx >> 11) & (DK - 1);
  int bh = idx >> 17;
  int b  = bh >> 4;
  int h  = bh & 15;
  vt[idx] = qkv[(size_t)(b * TSEQ + t) * F3 + 2 * CDIM + h * DK + d];
}

// ---------------------------------------------------------------------------
// bf16 WMMA GEMM:  C[M,N] = A[M,K] * B[N,K]^T   (both K-contiguous, row major)
// Wave tile 32x64: 2 A-frags x 4 B-frags -> 8 v_wmma per K=32 step.
// Block = 4 waves stacked in M -> 128x64 block tile.
// ---------------------------------------------------------------------------
template <bool OUT_BF16>
__global__ __launch_bounds__(128)
void gemm_bf16_wmma(const __bf16* __restrict__ A, const __bf16* __restrict__ B,
                    void* __restrict__ Cp, int K, int lda, int ldb, int ldc) {
  const int lane = threadIdx.x & 31;
  const int wave = threadIdx.x >> 5;
  const int half = lane >> 4;       // 0: lanes 0-15, 1: lanes 16-31
  const int l16  = lane & 15;
  const int m0 = blockIdx.x * 128 + wave * 32;
  const int n0 = blockIdx.y * 64;

  v8f acc[2][4];
#pragma unroll
  for (int i = 0; i < 2; ++i)
#pragma unroll
    for (int j = 0; j < 4; ++j)
      acc[i][j] = v8f{};

  for (int k0 = 0; k0 < K; k0 += 32) {
    Frag a[2], b[4];
#pragma unroll
    for (int i = 0; i < 2; ++i) {
      // A 16x32 frag: lane<16 -> K 0..15; lane>=16 -> K 16..31
      const __bf16* p = A + (size_t)(m0 + i * 16 + l16) * lda + k0 + half * 16;
      a[i].q[0] = *(const uint4*)p;
      a[i].q[1] = *(const uint4*)(p + 8);
    }
#pragma unroll
    for (int j = 0; j < 4; ++j) {
      // B 32x16 frag: lane<16 -> col n, K 0..15; lane>=16 -> K 16..31
      const __bf16* p = B + (size_t)(n0 + j * 16 + l16) * ldb + k0 + half * 16;
      b[j].q[0] = *(const uint4*)p;
      b[j].q[1] = *(const uint4*)(p + 8);
    }
#pragma unroll
    for (int i = 0; i < 2; ++i)
#pragma unroll
      for (int j = 0; j < 4; ++j)
        acc[i][j] = wmma_bf16(a[i], b[j], acc[i][j]);
  }

#pragma unroll
  for (int i = 0; i < 2; ++i)
#pragma unroll
    for (int j = 0; j < 4; ++j)
#pragma unroll
      for (int v = 0; v < 8; ++v) {
        const int row = m0 + i * 16 + v + half * 8;   // C layout: VGPR v = row v / v+8
        const int col = n0 + j * 16 + l16;
        if (OUT_BF16)
          ((__bf16*)Cp)[(size_t)row * ldc + col] = (__bf16)acc[i][j][v];
        else
          ((float*)Cp)[(size_t)row * ldc + col] = acc[i][j][v];
      }
}

// ---------------------------------------------------------------------------
// Flash attention (bf16 WMMA, fp32 accum, online softmax).
// grid = (B*H, T/64), block = 128 (4 waves); wave owns a 16-row Q tile and
// streams 64-key blocks:
//   8 wmma  : S(16x64) = Q * K^T
//   online softmax (row-max via ds_bpermute xor-reduce; row-SUM via an extra
//                   WMMA against an all-ones B fragment == virtual 65th col)
//   10 wmma : O(16x64) += P * V  and  l += P * 1
// ---------------------------------------------------------------------------
__global__ __launch_bounds__(128)
void attn_flash_wmma(const __bf16* __restrict__ qkv,
                     const __bf16* __restrict__ vt,
                     __bf16* __restrict__ aout) {
  __shared__ __align__(16) __bf16 lds_p[4][16][64];

  const int lane = threadIdx.x & 31;
  const int wave = threadIdx.x >> 5;
  const int half = lane >> 4;
  const int l16  = lane & 15;
  const int bh = blockIdx.x;
  const int b  = bh >> 4;
  const int h  = bh & 15;
  const int qt0 = blockIdx.y * 64 + wave * 16;   // query tile base in T

  const __bf16* Qbase = qkv + (size_t)(b * TSEQ) * F3 + h * DK;
  const __bf16* Kbase = Qbase + CDIM;
  const __bf16* Vt    = vt + (size_t)bh * DK * TSEQ;

  // Constant all-ones B fragment (bf16 1.0 = 0x3F80) for the row-sum column.
  Frag ones;
  ones.q[0] = make_uint4(0x3F803F80u, 0x3F803F80u, 0x3F803F80u, 0x3F803F80u);
  ones.q[1] = ones.q[0];

  // Q A-fragments for d = 0..31 and 32..63
  Frag aq[2];
#pragma unroll
  for (int kk = 0; kk < 2; ++kk) {
    const __bf16* p = Qbase + (size_t)(qt0 + l16) * F3 + kk * 32 + half * 16;
    aq[kk].q[0] = *(const uint4*)p;
    aq[kk].q[1] = *(const uint4*)(p + 8);
  }

  float mrun[8];
  v8f acc[5];                       // [0..3] = O tiles, [4] = row-sum l
#pragma unroll
  for (int v = 0; v < 8; ++v) mrun[v] = -1e30f;
#pragma unroll
  for (int nn = 0; nn < 5; ++nn) acc[nn] = v8f{};

  for (int jb = 0; jb < TSEQ / 64; ++jb) {
    const int kb = jb * 64;

    // ---- S = Q * K^T : four 16x16 tiles over 64 keys ----
    v8f s[4];
#pragma unroll
    for (int n = 0; n < 4; ++n) {
      Frag bk0, bk1;
      const __bf16* p = Kbase + (size_t)(kb + n * 16 + l16) * F3 + half * 16;
      bk0.q[0] = *(const uint4*)p;
      bk0.q[1] = *(const uint4*)(p + 8);
      bk1.q[0] = *(const uint4*)(p + 32);
      bk1.q[1] = *(const uint4*)(p + 40);
      v8f z = v8f{};
      z = wmma_bf16(aq[0], bk0, z);
      s[n] = wmma_bf16(aq[1], bk1, z);
    }

    // ---- scale + row max (rows = v + 8*half, cols across 16-lane group) ----
    float rm[8];
#pragma unroll
    for (int v = 0; v < 8; ++v) {
      s[0][v] *= 0.125f;            // 1/sqrt(64)
      s[1][v] *= 0.125f;
      s[2][v] *= 0.125f;
      s[3][v] *= 0.125f;
      rm[v] = fmaxf(fmaxf(s[0][v], s[1][v]), fmaxf(s[2][v], s[3][v]));
    }
#pragma unroll
    for (int off = 1; off <= 8; off <<= 1)
#pragma unroll
      for (int v = 0; v < 8; ++v)
        rm[v] = fmaxf(rm[v], __shfl_xor(rm[v], off, 32));

    float mnew[8], corr[8];
#pragma unroll
    for (int v = 0; v < 8; ++v) {
      mnew[v] = fmaxf(mrun[v], rm[v]);
      corr[v] = __expf(mrun[v] - mnew[v]);
      mrun[v] = mnew[v];
    }
#pragma unroll
    for (int n = 0; n < 4; ++n)
#pragma unroll
      for (int v = 0; v < 8; ++v)
        s[n][v] = __expf(s[n][v] - mnew[v]);
#pragma unroll
    for (int nn = 0; nn < 5; ++nn)
#pragma unroll
      for (int v = 0; v < 8; ++v) acc[nn][v] *= corr[v];

    // ---- C-layout P -> bf16 -> LDS (16x64 tile per wave) ----
#pragma unroll
    for (int n = 0; n < 4; ++n)
#pragma unroll
      for (int v = 0; v < 8; ++v)
        lds_p[wave][v + half * 8][n * 16 + l16] = (__bf16)s[n][v];
    __syncthreads();

    // ---- O += P * V ; l += P * 1   (two 32-key chunks) ----
#pragma unroll
    for (int kk = 0; kk < 2; ++kk) {
      Frag ap;
      const __bf16* lp = &lds_p[wave][l16][kk * 32 + half * 16];
      ap.q[0] = *(const uint4*)lp;
      ap.q[1] = *(const uint4*)(lp + 8);
#pragma unroll
      for (int nn = 0; nn < 4; ++nn) {
        Frag bv;
        const __bf16* p =
            Vt + (size_t)(nn * 16 + l16) * TSEQ + kb + kk * 32 + half * 16;
        bv.q[0] = *(const uint4*)p;
        bv.q[1] = *(const uint4*)(p + 8);
        acc[nn] = wmma_bf16(ap, bv, acc[nn]);
      }
      acc[4] = wmma_bf16(ap, ones, acc[4]);   // running softmax denominator
    }
    __syncthreads();
  }

  // ---- normalize by l (acc[4] holds row-sum in every column) and store ----
#pragma unroll
  for (int v = 0; v < 8; ++v) {
    const float inv = 1.0f / acc[4][v];
    const size_t g = (size_t)(b * TSEQ + qt0 + v + half * 8);
#pragma unroll
    for (int nn = 0; nn < 4; ++nn)
      aout[g * CDIM + h * DK + nn * 16 + l16] = (__bf16)(acc[nn][v] * inv);
  }
}

// ---------------------------------------------------------------------------
// Host-side launcher
// ---------------------------------------------------------------------------
extern "C" void kernel_launch(void* const* d_in, const int* in_sizes, int n_in,
                              void* d_out, int out_size, void* d_ws,
                              size_t ws_size, hipStream_t stream) {
  (void)in_sizes; (void)n_in; (void)out_size; (void)ws_size;
  const float* x     = (const float*)d_in[0];
  const float* w_qkv = (const float*)d_in[1];
  const float* w_o   = (const float*)d_in[2];
  float* out = (float*)d_out;

  char* ws = (char*)d_ws;
  const size_t MB = 1024ull * 1024ull;
  __bf16* xb    = (__bf16*)(ws + 0);        // 16 MB (reused as aout later)
  __bf16* wqkvb = (__bf16*)(ws + 16 * MB);  //  6 MB
  __bf16* wob   = (__bf16*)(ws + 22 * MB);  //  2 MB
  __bf16* qkvb  = (__bf16*)(ws + 24 * MB);  // 48 MB
  __bf16* vt    = (__bf16*)(ws + 72 * MB);  // 16 MB
  __bf16* aout  = xb;                        // x dead after GEMM1

  // 1) convert inputs to bf16
  {
    int n = MROWS * CDIM;
    f32_to_bf16_kernel<<<(n + 255) / 256, 256, 0, stream>>>(x, xb, n);
    n = F3 * CDIM;
    f32_to_bf16_kernel<<<(n + 255) / 256, 256, 0, stream>>>(w_qkv, wqkvb, n);
    n = CDIM * CDIM;
    f32_to_bf16_kernel<<<(n + 255) / 256, 256, 0, stream>>>(w_o, wob, n);
  }

  // 2) QKV projection: qkv[8192,3072] = xb[8192,1024] * w_qkv[3072,1024]^T
  gemm_bf16_wmma<true><<<dim3(MROWS / 128, F3 / 64), 128, 0, stream>>>(
      xb, wqkvb, qkvb, CDIM, CDIM, CDIM, F3);

  // 3) V transpose gather for PV fragments
  {
    int n = BATCH * NHEAD * DK * TSEQ;
    vt_gather_kernel<<<(n + 255) / 256, 256, 0, stream>>>(qkvb, vt);
  }

  // 4) flash attention
  attn_flash_wmma<<<dim3(BATCH * NHEAD, TSEQ / 64), 128, 0, stream>>>(
      qkvb, vt, aout);

  // 5) output projection: out[8192,1024] = aout * w_o^T  (f32 result)
  gemm_bf16_wmma<false><<<dim3(MROWS / 128, CDIM / 64), 128, 0, stream>>>(
      aout, wob, out, CDIM, CDIM, CDIM, CDIM);
}
